// GPT_84928683311448
// MI455X (gfx1250) — matrix operand order repaired and checked by reading, not compiled, not run on hardware
//
#include <hip/hip_runtime.h>

// GPT-2 small forward on gfx1250 (MI455X).
// - All GEMMs: v_wmma_f32_16x16x32_bf16, fp32 accumulate.
// - Weights converted+transposed to bf16 [N][K] once per launch; activations
//   produced in bf16 -> hot GEMM loops move raw bytes, no conversion VALU.
// - Double-buffered LDS tiles fed by GLOBAL_LOAD_ASYNC_TO_LDS_B128 +
//   s_wait_asynccnt (guarded; sync uint4 copy fallback).
// - Flash attention with transposed-score trick (softmax lane-local, P feeds
//   the P@V WMMA A-operand straight from the S^T accumulator).

#define DMODEL 768
#define NH     12
#define HSZ    64
#define TSEQ   1024
#define NBATCH 2
#define NLAYER 12
#define NROWS  (NBATCH * TSEQ)   // 2048
#define NVOC   50257

typedef __attribute__((ext_vector_type(16))) __bf16 bf16x16;
typedef __attribute__((ext_vector_type(8)))  float  f32x8;
typedef int v4i_ __attribute__((vector_size(16)));

union AFrag { bf16x16 v; uint4 q[2]; };

__device__ __forceinline__ unsigned short f2bf(float f) {
    unsigned int u = __builtin_bit_cast(unsigned int, f);
    unsigned int r = u + 0x7FFFu + ((u >> 16) & 1u);   // round-to-nearest-even
    return (unsigned short)(r >> 16);
}

// ---- CDNA5 async global->LDS copy (16B per lane), guarded ------------------
#if __has_builtin(__builtin_amdgcn_global_load_async_to_lds_b128) && \
    __has_builtin(__builtin_amdgcn_s_wait_asynccnt)
#define HAVE_ASYNC 1
__device__ __forceinline__ void cp16(void* lds, const void* g) {
    __builtin_amdgcn_global_load_async_to_lds_b128(
        (__attribute__((address_space(1))) v4i_*)(g),
        (__attribute__((address_space(3))) v4i_*)(lds), 0, 0);
}
#define ASYNC_WAIT(n) __builtin_amdgcn_s_wait_asynccnt(n)
#else
#define HAVE_ASYNC 0
__device__ __forceinline__ void cp16(void* lds, const void* g) {
    *(uint4*)lds = *(const uint4*)g;   // sync fallback: vmem load + ds store
}
#define ASYNC_WAIT(n) ((void)0)
#endif

// ---------------------------------------------------------------------------
// Weight convert + transpose: in [L][K][N] f32  ->  out [L][N][K] bf16
// ---------------------------------------------------------------------------
__global__ __launch_bounds__(256) void convt_kernel(
    const float* __restrict__ in, unsigned short* __restrict__ out, int K, int N)
{
    __shared__ float t[32][33];
    const size_t slice = (size_t)blockIdx.z * K * N;
    const int k0 = blockIdx.y * 32, n0 = blockIdx.x * 32;
    for (int i = threadIdx.y; i < 32; i += 8)
        t[i][threadIdx.x] = in[slice + (size_t)(k0 + i) * N + n0 + threadIdx.x];
    __syncthreads();
    for (int i = threadIdx.y; i < 32; i += 8)
        out[slice + (size_t)(n0 + i) * K + k0 + threadIdx.x] = f2bf(t[threadIdx.x][i]);
}

// Elementwise f32 -> bf16 (wte, already [N][K] for the tied head)
__global__ __launch_bounds__(256) void cvt_kernel(
    const float* __restrict__ in, unsigned short* __restrict__ out, long n4)
{
    const long i = (long)blockIdx.x * 256 + threadIdx.x;
    if (i >= n4) return;
    const float4 f = *(const float4*)(in + i * 4);
    out[i * 4 + 0] = f2bf(f.x); out[i * 4 + 1] = f2bf(f.y);
    out[i * 4 + 2] = f2bf(f.z); out[i * 4 + 3] = f2bf(f.w);
}

// ---------------------------------------------------------------------------
// Embedding: x[b,t,:] = wte[idx[b,t],:] + wpe[t,:]   (fp32 residual stream)
// ---------------------------------------------------------------------------
__global__ __launch_bounds__(256) void embed_kernel(
    const int* __restrict__ idx, const float* __restrict__ wte,
    const float* __restrict__ wpe, float* __restrict__ x)
{
    const int row = blockIdx.x;
    const int t   = row & (TSEQ - 1);
    const int id  = idx[row];
    const float* we = wte + (size_t)id * DMODEL;
    const float* wp = wpe + (size_t)t  * DMODEL;
    float* xr = x + (size_t)row * DMODEL;
    for (int c = threadIdx.x; c < DMODEL; c += 256)
        xr[c] = we[c] + wp[c];
}

// ---------------------------------------------------------------------------
// LayerNorm over D=768 (fp32 in, bf16 out), one block (8 waves) per row
// ---------------------------------------------------------------------------
__global__ __launch_bounds__(256) void ln_kernel(
    const float* __restrict__ x, const float* __restrict__ w,
    const float* __restrict__ b, unsigned short* __restrict__ y)
{
    __shared__ float s1[8], s2[8];
    const int row = blockIdx.x, tid = threadIdx.x;
    const float* xr = x + (size_t)row * DMODEL;
    float v[3], sum = 0.f, sq = 0.f;
#pragma unroll
    for (int i = 0; i < 3; i++) {
        v[i] = xr[tid + i * 256];
        sum += v[i]; sq += v[i] * v[i];
    }
#pragma unroll
    for (int o = 16; o > 0; o >>= 1) {
        sum += __shfl_xor(sum, o, 32);
        sq  += __shfl_xor(sq,  o, 32);
    }
    if ((tid & 31) == 0) { s1[tid >> 5] = sum; s2[tid >> 5] = sq; }
    __syncthreads();
    sum = 0.f; sq = 0.f;
#pragma unroll
    for (int i = 0; i < 8; i++) { sum += s1[i]; sq += s2[i]; }
    const float mu   = sum * (1.0f / DMODEL);
    const float var  = sq  * (1.0f / DMODEL) - mu * mu;
    const float rstd = rsqrtf(var + 1e-5f);
    unsigned short* yr = y + (size_t)row * DMODEL;
#pragma unroll
    for (int i = 0; i < 3; i++) {
        const int c = tid + i * 256;
        yr[c] = f2bf((v[i] - mu) * rstd * w[c] + b[c]);
    }
}

// ---------------------------------------------------------------------------
// bf16 WMMA GEMM: C[M,N] = A[M,K] @ Bt^T + bias (+GELU | +residual)
//   A:  [M][K] bf16 (activations); Bt: [N][K] bf16 (pre-transposed weights)
//   128x128x32 block tile, 8 waves (2x4), wave tile 64x32 (8 WMMA / k-step).
//   LDS: As[m][k] / Bs[n][k], stride 40 -> A-frag two 16B reads (k=8h, 8h+16),
//        B-frag the same two 16B reads (k=8h, 8h+16).  Double-buffered async copies.
//   EPI: 0=bias, 1=bias+GELU, 2=bias+residual(fp32).  OBF: bf16 output.
// ---------------------------------------------------------------------------
template<int EPI, bool OBF>
__global__ __launch_bounds__(256) void gemm_kernel(
    const unsigned short* __restrict__ A, const unsigned short* __restrict__ Bt,
    const float* __restrict__ bias, const float* __restrict__ res,
    void* __restrict__ Cout, int M, int N, int K)
{
    constexpr int BM = 128, BN = 128, BK = 32, LD = 40, NPT = 4;
    __shared__ alignas(16) unsigned short As[2][BM * LD];
    __shared__ alignas(16) unsigned short Bs[2][BN * LD];

    const int tid  = threadIdx.x;
    const int lane = tid & 31, wave = tid >> 5;
    const int hf   = lane >> 4, lml = lane & 15;
    const int wm   = wave >> 2, wn = wave & 3;
    const int m0   = blockIdx.y * BM, n0 = blockIdx.x * BN;

    // stage one 128x32 bf16 tile pair into buffer `buf` (4 async ops / wave)
    auto stage = [&](int k0, int buf) {
        for (int i = tid; i < 512; i += 256) {           // A: 128 rows x 4 chunks
            const int r = i >> 2, c = (i & 3) * 8;
            cp16(&As[buf][r * LD + c], A + (size_t)(m0 + r) * K + k0 + c);
        }
        for (int i = tid; i < 512; i += 256) {           // B: clamp OOB vocab rows
            const int r = i >> 2, c = (i & 3) * 8;
            const int gr = (n0 + r < N) ? (n0 + r) : (N - 1);
            cp16(&Bs[buf][r * LD + c], Bt + (size_t)gr * K + k0 + c);
        }
    };

    f32x8 acc[4][2] = {};
    const int niter = K / BK;
    stage(0, 0);

    for (int it = 0; it < niter; ++it) {
        const int cur = it & 1;
        __syncthreads();                       // all waves done reading buf cur^1
        if (it + 1 < niter) {
            stage((it + 1) * BK, cur ^ 1);
            ASYNC_WAIT(NPT);                   // own ops for tile `it` complete
        } else {
            ASYNC_WAIT(0);
        }
        __syncthreads();                       // tile `it` visible to all waves

        AFrag bfr[2];
#pragma unroll
        for (int b = 0; b < 2; b++) {
            const unsigned short* p = &Bs[cur][(wn * 32 + b * 16 + lml) * LD + hf * 8];
            bfr[b].q[0] = *(const uint4*)(p);
            bfr[b].q[1] = *(const uint4*)(p + 16);
        }
#pragma unroll
        for (int a = 0; a < 4; a++) {
            AFrag af;
            const unsigned short* p = &As[cur][(wm * 64 + a * 16 + lml) * LD + hf * 8];
            af.q[0] = *(const uint4*)(p);
            af.q[1] = *(const uint4*)(p + 16);
#pragma unroll
            for (int b = 0; b < 2; b++)
                acc[a][b] = __builtin_amdgcn_wmma_f32_16x16x32_bf16(
                    false, af.v, false, bfr[b].v, (short)0, acc[a][b], false, false);
        }
    }

    // epilogue: C-tile row M = r + 8*hf, col N = lml
#pragma unroll
    for (int a = 0; a < 4; a++) {
#pragma unroll
        for (int b = 0; b < 2; b++) {
            const int gn = n0 + wn * 32 + b * 16 + lml;
            if (gn >= N) continue;
            const float bv = bias ? bias[gn] : 0.0f;
#pragma unroll
            for (int r = 0; r < 8; r++) {
                const int gm = m0 + wm * 64 + a * 16 + hf * 8 + r;
                float v = acc[a][b][r] + bv;
                if (EPI == 1) v = 0.5f * v * (1.0f + erff(v * 0.70710678118654752f));
                if (EPI == 2) v += res[(size_t)gm * N + gn];
                if (OBF) ((unsigned short*)Cout)[(size_t)gm * N + gn] = f2bf(v);
                else     ((float*)Cout)[(size_t)gm * N + gn] = v;
            }
        }
    }
}

// ---------------------------------------------------------------------------
// Flash attention, causal, HS=64, qkv in bf16.  128 threads = 4 waves,
// 16 queries/wave.  S^T = K.Q^T keeps softmax lane-local; P fragment is
// built directly from the S^T accumulator (element pattern matches A-operand).
// ---------------------------------------------------------------------------
__global__ __launch_bounds__(128) void attn_kernel(
    const unsigned short* __restrict__ qkv, unsigned short* __restrict__ out)
{
    constexpr int LK = 72, LV = 40;
    __shared__ alignas(16) unsigned short Ks[32 * LK]; // K block [key][hs]
    __shared__ alignas(16) unsigned short Vt[64 * LV]; // V block [hs][key]

    const int tid  = threadIdx.x;
    const int lane = tid & 31, wave = tid >> 5;
    const int hf   = lane >> 4, lml = lane & 15;
    const int qb = blockIdx.x, hh = blockIdx.y, bb = blockIdx.z;
    const int qbase_wg = qb * 64;
    const int qbase    = qbase_wg + wave * 16;
    const int myq      = qbase + lml;
    const size_t rstr  = 3 * DMODEL;

    // Q fragments (B-operand): elem i <-> hs = 32j + 8h + i (i<8), 32j + 16 + 8h + i-8 (i>=8)  (raw bf16 copy)
    AFrag qf[2];
#pragma unroll
    for (int j = 0; j < 2; j++) {
        const unsigned short* qr =
            qkv + (size_t)(bb * TSEQ + myq) * rstr + hh * HSZ + j * 32 + hf * 8;
        qf[j].q[0] = *(const uint4*)(qr);
        qf[j].q[1] = *(const uint4*)(qr + 16);
    }

    f32x8 oacc[4] = {};
    float m = -1e30f, l = 0.0f;

    const int nblk = qbase_wg / 32 + 2;
    for (int kb = 0; kb < nblk; ++kb) {
        __syncthreads();
        for (int i = tid; i < 256; i += 128) {          // K: 32 rows x 8 chunks
            const int r = i >> 3, c = (i & 7) * 8;
            *(uint4*)(Ks + r * LK + c) = *(const uint4*)(qkv +
                (size_t)(bb * TSEQ + kb * 32 + r) * rstr + DMODEL + hh * HSZ + c);
        }
        for (int i = tid; i < 256; i += 128) {          // V: transpose to [hs][key]
            const int r = i >> 3, c = (i & 7) * 8;
            union { uint4 q; unsigned short s[8]; } u;
            u.q = *(const uint4*)(qkv +
                (size_t)(bb * TSEQ + kb * 32 + r) * rstr + 2 * DMODEL + hh * HSZ + c);
#pragma unroll
            for (int jj = 0; jj < 8; jj++) Vt[(c + jj) * LV + r] = u.s[jj];
        }
        __syncthreads();
        if (kb * 32 > qbase + 15) continue;             // fully-future block

        f32x8 st[2];
#pragma unroll
        for (int t = 0; t < 2; t++) {
            f32x8 s = {};
#pragma unroll
            for (int j = 0; j < 2; j++) {
                AFrag kf;
                const unsigned short* p = Ks + (t * 16 + lml) * LK + j * 32 + hf * 8;
                kf.q[0] = *(const uint4*)(p);
                kf.q[1] = *(const uint4*)(p + 16);
                s = __builtin_amdgcn_wmma_f32_16x16x32_bf16(
                    false, kf.v, false, qf[j].v, (short)0, s, false, false);
            }
            st[t] = s;
        }

        const bool diag = (kb * 32 + 31 > qbase);       // wave-uniform
        float rm = -1e30f;
#pragma unroll
        for (int t = 0; t < 2; t++)
#pragma unroll
            for (int r = 0; r < 8; r++) {
                float v = st[t][r] * 0.125f;            // 1/sqrt(64)
                if (diag) {
                    const int key = kb * 32 + t * 16 + hf * 8 + r;
                    if (key > myq) v = -1e30f;
                }
                st[t][r] = v;
                rm = fmaxf(rm, v);
            }
        rm = fmaxf(rm, __shfl_xor(rm, 16, 32));
        const float mnew  = fmaxf(m, rm);
        const float alpha = __expf(m - mnew);

        float rs = 0.0f;
        unsigned short ps[16];
#pragma unroll
        for (int t = 0; t < 2; t++)
#pragma unroll
            for (int r = 0; r < 8; r++) {
                const float p = __expf(st[t][r] - mnew);
                rs += p;
                ps[t * 8 + r] = f2bf(p);
            }
        rs += __shfl_xor(rs, 16, 32);
        l = l * alpha + rs;
        m = mnew;

#pragma unroll
        for (int r = 0; r < 8; r++) {                   // rescale O accumulators
            const float ar = __shfl(alpha, hf * 8 + r, 32);
#pragma unroll
            for (int j = 0; j < 4; j++) oacc[j][r] *= ar;
        }

        AFrag pf;                                       // P straight from S^T
        union { unsigned short s[16]; uint4 q[2]; } pu;
#pragma unroll
        for (int i = 0; i < 16; i++) pu.s[i] = ps[i];
        pf.q[0] = pu.q[0]; pf.q[1] = pu.q[1];
#pragma unroll
        for (int j = 0; j < 4; j++) {
            AFrag vf;
            const unsigned short* p = Vt + (j * 16 + lml) * LV + hf * 8;
            vf.q[0] = *(const uint4*)(p);
            vf.q[1] = *(const uint4*)(p + 16);
            oacc[j] = __builtin_amdgcn_wmma_f32_16x16x32_bf16(
                false, pf.v, false, vf.v, (short)0, oacc[j], false, false);
        }
    }

#pragma unroll
    for (int r = 0; r < 8; r++) {
        const float lr  = __shfl(l, hf * 8 + r, 32);
        const float inv = 1.0f / lr;
        const int q = qbase + hf * 8 + r;
        unsigned short* dst = out + (size_t)(bb * TSEQ + q) * DMODEL + hh * HSZ;
#pragma unroll
        for (int j = 0; j < 4; j++)
            dst[j * 16 + lml] = f2bf(oacc[j][r] * inv);
    }
}

// ---------------------------------------------------------------------------
extern "C" void kernel_launch(void* const* d_in, const int* in_sizes, int n_in,
                              void* d_out, int out_size, void* d_ws, size_t ws_size,
                              hipStream_t stream)
{
    (void)in_sizes; (void)n_in; (void)out_size; (void)ws_size;
    const int*   indices = (const int*)  d_in[0];
    const float* wte     = (const float*)d_in[1];
    const float* wpe     = (const float*)d_in[2];
    const float* ln1_w   = (const float*)d_in[3];
    const float* ln1_b   = (const float*)d_in[4];
    const float* wqkv    = (const float*)d_in[5];
    const float* bqkv    = (const float*)d_in[6];
    const float* wattn   = (const float*)d_in[7];
    const float* battn   = (const float*)d_in[8];
    const float* ln2_w   = (const float*)d_in[9];
    const float* ln2_b   = (const float*)d_in[10];
    const float* wfc     = (const float*)d_in[11];
    const float* bfc     = (const float*)d_in[12];
    const float* wmlp    = (const float*)d_in[13];
    const float* bmlp    = (const float*)d_in[14];
    const float* lnf_w   = (const float*)d_in[15];
    const float* lnf_b   = (const float*)d_in[16];
    float* out = (float*)d_out;

    // ---- workspace layout (~282 MB) ----
    char* p = (char*)d_ws;
    float* x = (float*)p;                 p += (size_t)NROWS * DMODEL * 4;
    unsigned short* h    = (unsigned short*)p; p += (size_t)NROWS * DMODEL * 2;
    unsigned short* qkv  = (unsigned short*)p; p += (size_t)NROWS * 3 * DMODEL * 2;
    unsigned short* att  = (unsigned short*)p; p += (size_t)NROWS * DMODEL * 2;
    unsigned short* mid  = (unsigned short*)p; p += (size_t)NROWS * 4 * DMODEL * 2;
    unsigned short* wqkv_t = (unsigned short*)p; p += (size_t)NLAYER * DMODEL * 3 * DMODEL * 2;
    unsigned short* wat_t  = (unsigned short*)p; p += (size_t)NLAYER * DMODEL * DMODEL * 2;
    unsigned short* wfc_t  = (unsigned short*)p; p += (size_t)NLAYER * DMODEL * 4 * DMODEL * 2;
    unsigned short* wmlp_t = (unsigned short*)p; p += (size_t)NLAYER * 4 * DMODEL * DMODEL * 2;
    unsigned short* wte_bf = (unsigned short*)p;

    // ---- weight preconversion (bf16, [N][K]) ----
    convt_kernel<<<dim3(3 * DMODEL / 32, DMODEL / 32, NLAYER), dim3(32, 8), 0, stream>>>(
        wqkv, wqkv_t, DMODEL, 3 * DMODEL);
    convt_kernel<<<dim3(DMODEL / 32, DMODEL / 32, NLAYER), dim3(32, 8), 0, stream>>>(
        wattn, wat_t, DMODEL, DMODEL);
    convt_kernel<<<dim3(4 * DMODEL / 32, DMODEL / 32, NLAYER), dim3(32, 8), 0, stream>>>(
        wfc, wfc_t, DMODEL, 4 * DMODEL);
    convt_kernel<<<dim3(DMODEL / 32, 4 * DMODEL / 32, NLAYER), dim3(32, 8), 0, stream>>>(
        wmlp, wmlp_t, 4 * DMODEL, DMODEL);
    {
        const long n4 = (long)NVOC * DMODEL / 4;        // 768 % 4 == 0
        cvt_kernel<<<(unsigned)((n4 + 255) / 256), 256, 0, stream>>>(wte, wte_bf, n4);
    }

    embed_kernel<<<NROWS, 256, 0, stream>>>(indices, wte, wpe, x);

    for (int l = 0; l < NLAYER; ++l) {
        ln_kernel<<<NROWS, 256, 0, stream>>>(x, ln1_w + l * DMODEL, ln1_b + l * DMODEL, h);
        gemm_kernel<0, true><<<dim3(3 * DMODEL / 128, NROWS / 128), 256, 0, stream>>>(
            h, wqkv_t + (size_t)l * DMODEL * 3 * DMODEL, bqkv + (size_t)l * 3 * DMODEL,
            nullptr, qkv, NROWS, 3 * DMODEL, DMODEL);
        attn_kernel<<<dim3(TSEQ / 64, NH, NBATCH), 128, 0, stream>>>(qkv, att);
        gemm_kernel<2, false><<<dim3(DMODEL / 128, NROWS / 128), 256, 0, stream>>>(
            att, wat_t + (size_t)l * DMODEL * DMODEL, battn + (size_t)l * DMODEL,
            x, x, NROWS, DMODEL, DMODEL);
        ln_kernel<<<NROWS, 256, 0, stream>>>(x, ln2_w + l * DMODEL, ln2_b + l * DMODEL, h);
        gemm_kernel<1, true><<<dim3(4 * DMODEL / 128, NROWS / 128), 256, 0, stream>>>(
            h, wfc_t + (size_t)l * DMODEL * 4 * DMODEL, bfc + (size_t)l * 4 * DMODEL,
            nullptr, mid, NROWS, 4 * DMODEL, DMODEL);
        gemm_kernel<2, false><<<dim3(DMODEL / 128, NROWS / 128), 256, 0, stream>>>(
            mid, wmlp_t + (size_t)l * 4 * DMODEL * DMODEL, bmlp + (size_t)l * DMODEL,
            x, x, NROWS, DMODEL, 4 * DMODEL);
    }

    ln_kernel<<<NROWS, 256, 0, stream>>>(x, lnf_w, lnf_b, h);
    gemm_kernel<0, false><<<dim3((NVOC + 127) / 128, NROWS / 128), 256, 0, stream>>>(
        h, wte_bf, nullptr, nullptr, out, NROWS, NVOC, DMODEL);
}
